// RescalMatcher_4269197492830
// MI455X (gfx1250) — matrix operand order repaired and checked by reading, not compiled, not run on hardware
//
#include <hip/hip_runtime.h>
#include <hip/hip_bf16.h>
#include <math.h>

typedef __attribute__((ext_vector_type(16))) _Float16 v16h;
typedef __attribute__((ext_vector_type(8)))  float    v8f;

#define D_    64
#define DM_   128
#define DI_   256
#define HID_  256
#define B_    128
#define NN_   200
#define FEW_  5
#define ROWS_ (B_ + FEW_)   /* 133 */
#define MPAD_ 144           /* 9 row-tiles of 16 */

__device__ __forceinline__ float sigf(float x) { return 1.0f / (1.0f + __expf(-x)); }

// ---------------------------------------------------------------------------
// K0: weight transpose + f32->f16 convert.  W (KxN, f32 row-major) ->
// WT (NxK, f16 row-major) so GEMM B-fragments become contiguous 32B loads.
// ---------------------------------------------------------------------------
__global__ void convertT_kernel(const float* __restrict__ W, _Float16* __restrict__ WT,
                                int K, int N)
{
    const int id = blockIdx.x * blockDim.x + threadIdx.x;
    if (id < K * N) {
        const int n = id / K, k = id % K;          // writes coalesced
        WT[id] = (_Float16)W[(size_t)k * N + n];
    }
}

// ---------------------------------------------------------------------------
// K1: neighbor encoder.  One block per (row, side).  256 threads:
// thread t -> output row i = t>>2, k-quarter q = t&3.  Private accumulator
// across all 200 neighbors; e-vector staged into LDS with
// GLOBAL_LOAD_ASYNC_TO_LDS_B128 (ASYNCcnt); one reduction at the end.
// gcn_w folds to a single 64x64 matvec: sum_n (R e) @ W == (sum_n R e) @ W.
// ---------------------------------------------------------------------------
__global__ __launch_bounds__(256) void neighbor_kernel(
    const int* __restrict__ qlc, const int* __restrict__ qrc,
    const int* __restrict__ slc, const int* __restrict__ src_,
    const float* __restrict__ qld, const float* __restrict__ qrd,
    const float* __restrict__ sld, const float* __restrict__ srd,
    const float* __restrict__ ent, const float* __restrict__ rel,
    const float* __restrict__ gw,  const float* __restrict__ gwb,
    const float* __restrict__ gb,  float* __restrict__ X)
{
    const int row  = blockIdx.x;   // 0..132 (0..127 query, 128..132 support)
    const int side = blockIdx.y;   // 0 = left, 1 = right
    const int t = threadIdx.x;     // 0..255
    const int i = t >> 2;          // 0..63 output component
    const int q = t & 3;           // k-quarter (16 elems each)

    const int* conn; float deg;
    if (row < B_) {
        conn = (side ? qrc : qlc) + row * NN_ * 2;
        deg  = (side ? qrd : qld)[row];
    } else {
        const int s = row - B_;
        conn = (side ? src_ : slc) + s * NN_ * 2;
        deg  = (side ? srd : sld)[s];
    }

    __shared__ __align__(16) float ldsE[64];
    __shared__ float red[256];
    __shared__ float msum[64];

    // LDS byte offset of ldsE: LDS aperture maps flat addr[31:0] -> LDS offset
    const unsigned ldsBase = (unsigned)(size_t)(void*)ldsE;

    float acc = 0.f;
    for (int n = 0; n < NN_; ++n) {
        const int r  = conn[2 * n + 0];
        const int ei = conn[2 * n + 1];
        if (t < 16) {
            // async DMA 16B per lane: ent row (256B) -> LDS, no VGPR round-trip
            const void* g = (const char*)(ent + (size_t)ei * D_) + t * 16;
            const unsigned l = ldsBase + t * 16;
            asm volatile("global_load_async_to_lds_b128 %0, %1, off"
                         :: "v"(l), "v"(g) : "memory");
        }
        asm volatile("s_wait_asynccnt 0x0" ::: "memory");
        __syncthreads();

        const float* Rp = rel + (size_t)r * (D_ * D_) + i * D_ + q * 16;
        const float4 r0 = ((const float4*)Rp)[0];
        const float4 r1 = ((const float4*)Rp)[1];
        const float4 r2 = ((const float4*)Rp)[2];
        const float4 r3 = ((const float4*)Rp)[3];
        const float4 e0 = ((const float4*)ldsE)[q * 4 + 0];
        const float4 e1 = ((const float4*)ldsE)[q * 4 + 1];
        const float4 e2 = ((const float4*)ldsE)[q * 4 + 2];
        const float4 e3 = ((const float4*)ldsE)[q * 4 + 3];
        acc += r0.x*e0.x + r0.y*e0.y + r0.z*e0.z + r0.w*e0.w
             + r1.x*e1.x + r1.y*e1.y + r1.z*e1.z + r1.w*e1.w
             + r2.x*e2.x + r2.y*e2.y + r2.z*e2.z + r2.w*e2.w
             + r3.x*e3.x + r3.y*e3.y + r3.z*e3.z + r3.w*e3.w;

        if (n + 1 < NN_)  // pull next relation tile toward L0/L2 (global_prefetch_b8)
            __builtin_prefetch(rel + (size_t)conn[2*(n+1)] * (D_*D_) + i * D_ + q * 16, 0, 0);
        __syncthreads();
    }

    red[t] = acc;
    __syncthreads();
    if (t < 64) msum[t] = red[4*t] + red[4*t+1] + red[4*t+2] + red[4*t+3];
    __syncthreads();
    if (t < 64) {
        float o = 0.f;
        #pragma unroll 8
        for (int ii = 0; ii < 64; ++ii) o += msum[ii] * gw[ii * 64 + t];
        o += (float)NN_ * (gwb[t] + gb[t]);
        o = tanhf(o / deg);
        X[(size_t)row * DM_ + side * D_ + t] = o;
    }
}

// ---------------------------------------------------------------------------
// K2: specialized f16-in / f32-accumulate WMMA GEMM.
// C = op(A) @ Bt^T (+bias)(+bias2)(+resid).  One wave per 16x16 C tile,
// K = KCH*32 fully unrolled -> KCH back-to-back v_wmma_f32_16x16x32_f16.
// A,C f32 row-major; Bt is pre-converted f16, N-major (Bt[n*K + k]), so each
// lane's B fragment is one contiguous 32B (v16h) load.
// ---------------------------------------------------------------------------
template<int KCH, bool RELU, bool HASB, bool HASB2, bool HASR, bool GUARDM>
__global__ __launch_bounds__(256) void wmma_gemm(
    const float* __restrict__ A, const _Float16* __restrict__ Bt,
    float* __restrict__ C, int M, int N,
    int lda, int ldc,
    const float* __restrict__ bias, const float* __restrict__ bias2,
    const float* __restrict__ resid, int ldr)
{
    constexpr int K = KCH * 32;
    const int lane = threadIdx.x & 31;
    const int wave = threadIdx.x >> 5;
    const int nT = N >> 4;
    const int mT = (M + 15) >> 4;
    const int tile = blockIdx.x * (blockDim.x >> 5) + wave;
    if (tile >= mT * nT) return;                 // wave-uniform: EXEC all-1 at WMMA
    const int tm = tile / nT, tn = tile % nT;
    const int h  = lane >> 4;                    // lane half
    const int lr = lane & 15;
    int arow = tm * 16 + lr;
    if (GUARDM) { if (arow > M - 1) arow = M - 1; }
    const int n = tn * 16 + lr;

    v8f acc = {0.f, 0.f, 0.f, 0.f, 0.f, 0.f, 0.f, 0.f};
    #pragma unroll
    for (int kc = 0; kc < KCH; ++kc) {
        const int k0 = kc * 32;
        // B 32x16 f16, this code's contiguous k order: lanes 0-15 hold K=0..15, lanes 16-31 K=16..31;
        // contiguous in Bt -> single v16h (2x b128) load per chunk.
        const v16h b = *(const v16h*)(Bt + (size_t)n * K + k0 + h * 16);
        v16h a;
        #pragma unroll
        for (int j = 0; j < 16; ++j) {
            // A 16x32 f16, same contiguous k order: element j -> K=h*16+j (matches B)
            const int ka = k0 + h * 16 + j;
            float av = A[(size_t)arow * lda + ka];
            if (RELU) av = fmaxf(av, 0.f);
            a[j] = (_Float16)av;
        }
        acc = __builtin_amdgcn_wmma_f32_16x16x32_f16(false, a, false, b,
                                                     (short)0, acc, false, false);
    }

    float bn = 0.f;
    if (HASB)  bn += bias[n];
    if (HASB2) bn += bias2[n];

    #pragma unroll
    for (int v = 0; v < 8; ++v) {
        const int m = tm * 16 + v + 8 * h;       // C layout: VGPR v -> M = v + 8*half
        if (!GUARDM || m < M) {
            float val = acc[v] + bn;
            if (HASR) val += resid[(size_t)m * ldr + n];
            C[(size_t)m * ldc + n] = val;
        }
    }
}

// ---------------------------------------------------------------------------
// K3: layernorm with ddof=1 std, per row of 128.  133 blocks x 128 threads.
// ---------------------------------------------------------------------------
__global__ void ln_kernel(const float* __restrict__ HP,
                          const float* __restrict__ g, const float* __restrict__ b,
                          float* __restrict__ G)
{
    const int row = blockIdx.x, t = threadIdx.x;
    __shared__ float sb[128];
    const float hv = HP[(size_t)row * DM_ + t];
    sb[t] = hv; __syncthreads();
    for (int s = 64; s > 0; s >>= 1) { if (t < s) sb[t] += sb[t + s]; __syncthreads(); }
    const float mu = sb[0] * (1.0f / 128.0f);
    __syncthreads();
    const float d = hv - mu;
    sb[t] = d * d; __syncthreads();
    for (int s = 64; s > 0; s >>= 1) { if (t < s) sb[t] += sb[t + s]; __syncthreads(); }
    const float sigma = sqrtf(sb[0] * (1.0f / 127.0f));
    G[(size_t)row * DM_ + t] = d / (sigma + 0.001f) * g[t] + b[t];
}

// ---------------------------------------------------------------------------
// K4: LSTM cell nonlinearity + h update.  128 blocks x 256 threads.
// gates layout: [i | f | g | o] each 256 wide.  h_r[:, :128] = query_g + h_r_[:, :128]
// ---------------------------------------------------------------------------
__global__ void lstm_cell(const float* __restrict__ gates, float* __restrict__ c,
                          const float* __restrict__ qg, float* __restrict__ h_r)
{
    const int row = blockIdx.x, j = threadIdx.x;
    const float* gr = gates + (size_t)row * 1024;
    const float ig = gr[j], fg = gr[256 + j], gg = gr[512 + j], og = gr[768 + j];
    const float cn = sigf(fg) * c[(size_t)row * HID_ + j] + sigf(ig) * tanhf(gg);
    c[(size_t)row * HID_ + j] = cn;
    const float hro = sigf(og) * tanhf(cn);
    if (j < DM_) h_r[(size_t)row * HID_ + j] = qg[(size_t)row * DM_ + j] + hro;
}

// ---------------------------------------------------------------------------
// K5: per-row attention over 5 supports; writes r into h_r[:, 128:256].
// ---------------------------------------------------------------------------
__global__ void attn_kernel(float* __restrict__ h_r, const float* __restrict__ supp)
{
    const int row = blockIdx.x, t = threadIdx.x;   // 128 threads
    __shared__ float a[FEW_];
    const float* hv = h_r + (size_t)row * HID_;
    if (t < FEW_) {
        float s = 0.f;
        #pragma unroll 8
        for (int j = 0; j < DM_; ++j) s += hv[j] * supp[(size_t)t * DM_ + j];
        a[t] = s;
    }
    __syncthreads();
    if (t == 0) {
        float m = a[0];
        #pragma unroll
        for (int s = 1; s < FEW_; ++s) m = fmaxf(m, a[s]);
        float z = 0.f;
        #pragma unroll
        for (int s = 0; s < FEW_; ++s) { a[s] = __expf(a[s] - m); z += a[s]; }
        const float inv = 1.f / z;
        #pragma unroll
        for (int s = 0; s < FEW_; ++s) a[s] *= inv;
    }
    __syncthreads();
    float r = 0.f;
    #pragma unroll
    for (int s = 0; s < FEW_; ++s) r += a[s] * supp[(size_t)s * DM_ + t];
    h_r[(size_t)row * HID_ + DM_ + t] = r;
}

// ---------------------------------------------------------------------------
// K6: final scores out = h @ support_g.T  (128 x 5)
// ---------------------------------------------------------------------------
__global__ void out_kernel(const float* __restrict__ h_r, const float* __restrict__ supp,
                           float* __restrict__ out)
{
    const int row = blockIdx.x, t = threadIdx.x;   // 32 threads, use 5
    if (t < FEW_) {
        const float* hv = h_r + (size_t)row * HID_;
        float s = 0.f;
        #pragma unroll 8
        for (int j = 0; j < DM_; ++j) s += hv[j] * supp[(size_t)t * DM_ + j];
        out[row * FEW_ + t] = s;
    }
}

__global__ void zero_kernel(float* __restrict__ p, int n)
{
    const int idx = blockIdx.x * blockDim.x + threadIdx.x;
    if (idx < n) p[idx] = 0.f;
}

// ---------------------------------------------------------------------------
extern "C" void kernel_launch(void* const* d_in, const int* in_sizes, int n_in,
                              void* d_out, int out_size, void* d_ws, size_t ws_size,
                              hipStream_t stream)
{
    const int*   qlc = (const int*)  d_in[0];
    const int*   qrc = (const int*)  d_in[1];
    const int*   slc = (const int*)  d_in[2];
    const int*   srcn= (const int*)  d_in[3];
    const float* qld = (const float*)d_in[4];
    const float* qrd = (const float*)d_in[5];
    const float* sld = (const float*)d_in[6];
    const float* srd = (const float*)d_in[7];
    const float* ent = (const float*)d_in[8];
    const float* rel = (const float*)d_in[9];
    const float* gw  = (const float*)d_in[10];
    const float* gwb = (const float*)d_in[11];
    const float* gb  = (const float*)d_in[12];
    const float* w1  = (const float*)d_in[13];
    const float* b1  = (const float*)d_in[14];
    const float* w2  = (const float*)d_in[15];
    const float* b2  = (const float*)d_in[16];
    const float* lng = (const float*)d_in[17];
    const float* lnb = (const float*)d_in[18];
    const float* wih = (const float*)d_in[19];
    const float* whh = (const float*)d_in[20];
    const float* bih = (const float*)d_in[21];
    const float* bhh = (const float*)d_in[22];
    float* out = (float*)d_out;

    // workspace layout (f32 then f16)
    float* ws    = (float*)d_ws;
    float* X     = ws;                     // MPAD_ x 128
    float* H1    = X     + MPAD_ * DM_;    // MPAD_ x 256
    float* HP    = H1    + MPAD_ * DI_;    // MPAD_ x 128
    float* G     = HP    + MPAD_ * DM_;    // MPAD_ x 128 (query_g rows 0..127, support_g 128..132)
    float* xproj = G     + MPAD_ * DM_;    // 128 x 1024
    float* gates = xproj + B_ * 1024;      // 128 x 1024
    float* h_r   = gates + B_ * 1024;      // 128 x 256
    float* cst   = h_r   + B_ * HID_;      // 128 x 256 (contiguous with h_r)
    float* suppG = G + B_ * DM_;           // support_g

    _Float16* w1T  = (_Float16*)(cst + B_ * HID_);  // 256 x 128
    _Float16* w2T  = w1T  + DI_ * DM_;              // 128 x 256
    _Float16* wihT = w2T  + DM_ * DI_;              // 1024 x 128
    _Float16* whhT = wihT + 1024 * DM_;             // 1024 x 256

    // 0) weight transpose + f16 convert (tiny one-time cost; enables 32B B-fragments)
    convertT_kernel<<<(DM_ * DI_    + 255) / 256, 256, 0, stream>>>(w1,  w1T,  DM_,  DI_);
    convertT_kernel<<<(DI_ * DM_    + 255) / 256, 256, 0, stream>>>(w2,  w2T,  DI_,  DM_);
    convertT_kernel<<<(DM_ * 1024   + 255) / 256, 256, 0, stream>>>(wih, wihT, DM_,  1024);
    convertT_kernel<<<(HID_ * 1024  + 255) / 256, 256, 0, stream>>>(whh, whhT, HID_, 1024);

    // 1) neighbor encoder -> X (133 x 128)
    neighbor_kernel<<<dim3(ROWS_, 2), 256, 0, stream>>>(
        qlc, qrc, slc, srcn, qld, qrd, sld, srd, ent, rel, gw, gwb, gb, X);

    // 2) H1 = X @ w1 + b1              (M=133, N=256, K=128)
    {
        int tiles = ((ROWS_ + 15) / 16) * (DI_ / 16);
        wmma_gemm<4, false, true, false, false, true><<<(tiles + 7) / 8, 256, 0, stream>>>(
            X, w1T, H1, ROWS_, DI_, DM_, DI_, b1, nullptr, nullptr, 0);
    }
    // 3) HP = relu(H1) @ w2 + b2 + X   (M=133, N=128, K=256)
    {
        int tiles = ((ROWS_ + 15) / 16) * (DM_ / 16);
        wmma_gemm<8, true, true, false, true, true><<<(tiles + 7) / 8, 256, 0, stream>>>(
            H1, w2T, HP, ROWS_, DM_, DI_, DM_, b2, nullptr, X, DM_);
    }
    // 4) layernorm -> G
    ln_kernel<<<ROWS_, 128, 0, stream>>>(HP, lng, lnb, G);

    // 5) x_proj = query_g @ wih + bih + bhh  (M=128, N=1024, K=128)
    {
        int tiles = (B_ / 16) * (4 * HID_ / 16);
        wmma_gemm<4, false, true, true, false, false><<<(tiles + 7) / 8, 256, 0, stream>>>(
            G, wihT, xproj, B_, 4 * HID_, DM_, 4 * HID_, bih, bhh, nullptr, 0);
    }

    // 6) zero h_r and c (contiguous 2 * 128 * 256 floats)
    zero_kernel<<<(2 * B_ * HID_ + 255) / 256, 256, 0, stream>>>(h_r, 2 * B_ * HID_);

    // 7) 4 recurrent steps
    for (int step = 0; step < 4; ++step) {
        // gates = h_r @ whh + x_proj   (M=128, N=1024, K=256)
        int tiles = (B_ / 16) * (4 * HID_ / 16);
        wmma_gemm<8, false, false, false, true, false><<<(tiles + 7) / 8, 256, 0, stream>>>(
            h_r, whhT, gates, B_, 4 * HID_, HID_, 4 * HID_,
            nullptr, nullptr, xproj, 4 * HID_);
        lstm_cell<<<B_, 256, 0, stream>>>(gates, cst, G, h_r);
        attn_kernel<<<B_, 128, 0, stream>>>(h_r, suppG);
    }

    // 8) out = h @ support_g.T
    out_kernel<<<B_, 32, 0, stream>>>(h_r, suppG, out);
}
